// PerceiverAR_26946624815765
// MI455X (gfx1250) — compile-verified, matrix operand order checked by reading, not run on hardware
//
#include <hip/hip_runtime.h>

typedef __attribute__((ext_vector_type(16))) __bf16 v16bf;
typedef __attribute__((ext_vector_type(8)))  __bf16 v8bf;
typedef __attribute__((ext_vector_type(8)))  float  v8f;
typedef __attribute__((ext_vector_type(4)))  int    i32x4;

#if __has_builtin(__builtin_amdgcn_global_load_async_to_lds_b128)
#define HAS_ASYNC_LDS 1
#else
#define HAS_ASYNC_LDS 0
#endif

namespace cfg {
constexpr int B = 4, S = 8192, Z = 1024, C = 1024, HS = 8, L = 2, NB = 2;
}

enum { MODE_BF16 = 0, MODE_SQRELU = 1, MODE_F32ADD = 2, MODE_SCORES = 3, MODE_VT = 4 };

__device__ __forceinline__ void wait_async0() {
#if HAS_ASYNC_LDS
#if __has_builtin(__builtin_amdgcn_s_wait_asynccnt)
  __builtin_amdgcn_s_wait_asynccnt(0);
#else
  asm volatile("s_wait_asynccnt 0x0" ::: "memory");
#endif
#endif
}

// ---------------------------------------------------------------------------
// fp32 [K][N] -> bf16 [N][K]  (so GEMM B-operand is K-contiguous per column)
// ---------------------------------------------------------------------------
__global__ void k_transpose_bf16(const float* __restrict__ W, __bf16* __restrict__ Wt,
                                 int K, int N) {
  long t = (long)blockIdx.x * blockDim.x + threadIdx.x;
  if (t >= (long)K * N) return;
  int n = (int)(t / K), k = (int)(t % K);
  Wt[t] = (__bf16)W[(long)k * N + n];
}

// ---------------------------------------------------------------------------
// Embedding gather + two LayerNorms (kv-LN over all S rows; q-LN + raw copy
// for the last Z rows which form the latent residual stream z).
// ---------------------------------------------------------------------------
__global__ __launch_bounds__(256)
void k_gather_ln(const int* __restrict__ inputs, const float* __restrict__ emb,
                 const float* __restrict__ g_kv, const float* __restrict__ b_kv,
                 const float* __restrict__ g_q,  const float* __restrict__ b_q,
                 __bf16* __restrict__ en, float* __restrict__ z,
                 __bf16* __restrict__ zn) {
  using namespace cfg;
  __shared__ float sh[256];
  __shared__ float sh2[256];
  int row = blockIdx.x;                 // 0 .. B*S-1
  int b = row / S, s = row % S;
  int t = threadIdx.x;
  const float* e = emb + (long)inputs[row] * C;
  float x[4];
  float sum = 0.f, sq = 0.f;
  for (int j = 0; j < 4; ++j) { float v = e[t + j * 256]; x[j] = v; sum += v; sq += v * v; }
  sh[t] = sum; sh2[t] = sq; __syncthreads();
  for (int o = 128; o > 0; o >>= 1) {
    if (t < o) { sh[t] += sh[t + o]; sh2[t] += sh2[t + o]; }
    __syncthreads();
  }
  float mean = sh[0] * (1.f / C);
  float var  = sh2[0] * (1.f / C) - mean * mean;
  float rstd = rsqrtf(var + 1e-5f);
  long eo = (long)row * C;
  for (int j = 0; j < 4; ++j) {
    int i = t + j * 256;
    float nv = (x[j] - mean) * rstd;
    en[eo + i] = (__bf16)(nv * g_kv[i] + b_kv[i]);
  }
  if (s >= S - Z) {
    long zr = (long)(b * Z + (s - (S - Z))) * C;
    for (int j = 0; j < 4; ++j) {
      int i = t + j * 256;
      float nv = (x[j] - mean) * rstd;
      z[zr + i]  = x[j];
      zn[zr + i] = (__bf16)(nv * g_q[i] + b_q[i]);
    }
  }
}

// ---------------------------------------------------------------------------
// LayerNorm over fp32 residual -> bf16 (one block per row, C = 1024)
// ---------------------------------------------------------------------------
__global__ __launch_bounds__(256)
void k_ln(const float* __restrict__ xg, const float* __restrict__ g,
          const float* __restrict__ bta, __bf16* __restrict__ out) {
  using namespace cfg;
  __shared__ float sh[256];
  __shared__ float sh2[256];
  int t = threadIdx.x;
  long ro = (long)blockIdx.x * C;
  float x[4];
  float sum = 0.f, sq = 0.f;
  for (int j = 0; j < 4; ++j) { float v = xg[ro + t + j * 256]; x[j] = v; sum += v; sq += v * v; }
  sh[t] = sum; sh2[t] = sq; __syncthreads();
  for (int o = 128; o > 0; o >>= 1) {
    if (t < o) { sh[t] += sh[t + o]; sh2[t] += sh2[t + o]; }
    __syncthreads();
  }
  float mean = sh[0] * (1.f / C);
  float rstd = rsqrtf(sh2[0] * (1.f / C) - mean * mean + 1e-5f);
  for (int j = 0; j < 4; ++j) {
    int i = t + j * 256;
    out[ro + i] = (__bf16)((x[j] - mean) * rstd * g[i] + bta[i]);
  }
}

// ---------------------------------------------------------------------------
// Row softmax: fp32 scores -> bf16 probabilities. One block per row.
// ---------------------------------------------------------------------------
__global__ __launch_bounds__(256)
void k_softmax(const float* __restrict__ sc, __bf16* __restrict__ p, int N) {
  __shared__ float sh[256];
  long ro = (long)blockIdx.x * N;
  int t = threadIdx.x;
  float m = -3.0e38f;
  for (int i = t; i < N; i += 256) m = fmaxf(m, sc[ro + i]);
  sh[t] = m; __syncthreads();
  for (int o = 128; o > 0; o >>= 1) { if (t < o) sh[t] = fmaxf(sh[t], sh[t + o]); __syncthreads(); }
  m = sh[0]; __syncthreads();
  float sum = 0.f;
  for (int i = t; i < N; i += 256) sum += __expf(sc[ro + i] - m);
  sh[t] = sum; __syncthreads();
  for (int o = 128; o > 0; o >>= 1) { if (t < o) sh[t] += sh[t + o]; __syncthreads(); }
  float inv = 1.f / sh[0];
  for (int i = t; i < N; i += 256) p[ro + i] = (__bf16)(__expf(sc[ro + i] - m) * inv);
}

// ---------------------------------------------------------------------------
// Unified bf16 WMMA GEMM (double-buffered async LDS pipeline on gfx1250).
//   A: bf16 [M][K] row-major (lda), per-z: + zb*aZ + zh*aH
//   B: bf16 [N][K] row-major (ldb) -> column n of math-B is K-contiguous
// Block: 256 threads = 8 wave32s; tile 128(M) x 64(N), K-tile 64.
// Each wave: 32x32 via 2x2 v_wmma_f32_16x16x32_bf16 fragments.
// ---------------------------------------------------------------------------
__global__ __launch_bounds__(256)
void k_gemm_wmma(const __bf16* __restrict__ A, long aZ, long aH, int lda,
                 const __bf16* __restrict__ Bm, long bZ, long bH, int ldb,
                 const float* __restrict__ bias,
                 void* __restrict__ OutV, long oZ, long oH, int ldo,
                 int K, int H, int mode, float scale, int maskOff,
                 int Srows, long vtBatch) {
  // padded rows: 72 elems = 144 B (16B aligned, bank-conflict-free fragment reads)
  __shared__ __bf16 As[2][128 * 72];
  __shared__ __bf16 Bs[2][64 * 72];

  const int zb = (int)blockIdx.z / H, zh = (int)blockIdx.z % H;
  const __bf16* Ap = A  + (long)zb * aZ + (long)zh * aH + (long)blockIdx.y * 128 * lda;
  const __bf16* Bp = Bm + (long)zb * bZ + (long)zh * bH + (long)blockIdx.x * 64  * ldb;

  const int t = threadIdx.x;
  const int wid = t >> 5, l = t & 31, half = l >> 4, lm = l & 15;
  const int wm = (wid & 3) * 32;    // wave M offset within block tile
  const int wn = (wid >> 2) * 32;   // wave N offset within block tile

  v8f acc[2][2] = {};

  auto compute_tile = [&](int buf) {
    #pragma unroll
    for (int s = 0; s < 2; ++s) {
      const int ks = s * 32;
      v16bf af[2], bfv[2];
      #pragma unroll
      for (int mi = 0; mi < 2; ++mi) {
        const __bf16* pa = &As[buf][(wm + mi * 16 + lm) * 72 + ks + half * 8];
        v8bf lo = *(const v8bf*)pa;
        v8bf hi = *(const v8bf*)(pa + 16);
        af[mi] = __builtin_shufflevector(lo, hi, 0,1,2,3,4,5,6,7,8,9,10,11,12,13,14,15);
      }
      #pragma unroll
      for (int ni = 0; ni < 2; ++ni) {
        const __bf16* pb = &Bs[buf][(wn + ni * 16 + lm) * 72 + ks + half * 8];
        v8bf lo = *(const v8bf*)pb;
        v8bf hi = *(const v8bf*)(pb + 16);
        bfv[ni] = __builtin_shufflevector(lo, hi, 0,1,2,3,4,5,6,7,8,9,10,11,12,13,14,15);
      }
      #pragma unroll
      for (int mi = 0; mi < 2; ++mi)
        #pragma unroll
        for (int ni = 0; ni < 2; ++ni)
          acc[mi][ni] = __builtin_amdgcn_wmma_f32_16x16x32_bf16(
              false, af[mi], false, bfv[ni], (short)0, acc[mi][ni], false, false);
    }
  };

#if HAS_ASYNC_LDS
  // ---- CDNA5 async global->LDS pipeline (ASYNCcnt-tracked, zero VGPR staging)
  auto load_tile = [&](int buf, int kk) {
    #pragma unroll
    for (int i = 0; i < 4; ++i) {              // A: 128x64 bf16 = 1024 16B chunks
      int cid = t + i * 256;
      int r = cid >> 3, c = cid & 7;
      const __bf16* gp = Ap + (long)r * lda + kk + c * 8;
      __bf16* lp = &As[buf][r * 72 + c * 8];
      __builtin_amdgcn_global_load_async_to_lds_b128(
          (__attribute__((address_space(1))) i32x4*)gp,
          (__attribute__((address_space(3))) i32x4*)lp, 0, 0);
    }
    #pragma unroll
    for (int i = 0; i < 2; ++i) {              // B: 64x64 bf16 = 512 16B chunks
      int cid = t + i * 256;
      int r = cid >> 3, c = cid & 7;
      const __bf16* gp = Bp + (long)r * ldb + kk + c * 8;
      __bf16* lp = &Bs[buf][r * 72 + c * 8];
      __builtin_amdgcn_global_load_async_to_lds_b128(
          (__attribute__((address_space(1))) i32x4*)gp,
          (__attribute__((address_space(3))) i32x4*)lp, 0, 0);
    }
  };

  const int nk = K >> 6;
  load_tile(0, 0);
  for (int it = 0; it < nk; ++it) {
    wait_async0();        // this wave's copies for tile `it` have landed
    __syncthreads();      // everyone's copies landed; prev buffer's readers done
    if (it + 1 < nk) load_tile((it + 1) & 1, (it + 1) << 6);
    compute_tile(it & 1);
  }
#else
  // ---- fallback: batched register staging (issue all loads, then all stores)
  for (int kk = 0; kk < K; kk += 64) {
    __syncthreads();
    uint4 ta[4], tb[2];
    #pragma unroll
    for (int i = 0; i < 4; ++i) {
      int cid = t + i * 256;
      int r = cid >> 3, c = cid & 7;
      ta[i] = *(const uint4*)(Ap + (long)r * lda + kk + c * 8);
    }
    #pragma unroll
    for (int i = 0; i < 2; ++i) {
      int cid = t + i * 256;
      int r = cid >> 3, c = cid & 7;
      tb[i] = *(const uint4*)(Bp + (long)r * ldb + kk + c * 8);
    }
    #pragma unroll
    for (int i = 0; i < 4; ++i) {
      int cid = t + i * 256;
      int r = cid >> 3, c = cid & 7;
      *(uint4*)(&As[0][r * 72 + c * 8]) = ta[i];
    }
    #pragma unroll
    for (int i = 0; i < 2; ++i) {
      int cid = t + i * 256;
      int r = cid >> 3, c = cid & 7;
      *(uint4*)(&Bs[0][r * 72 + c * 8]) = tb[i];
    }
    __syncthreads();
    compute_tile(0);
  }
#endif

  // ---- epilogue ----
  const int mBase = (int)blockIdx.y * 128 + wm + half * 8;  // D layout: m = r + 8*(lane/16)
  const int nBase = (int)blockIdx.x * 64 + wn + lm;         //           n = lane%16

  if (mode == MODE_F32ADD) {
    float* O = (float*)OutV + (long)zb * oZ + (long)zh * oH;
    #pragma unroll
    for (int mi = 0; mi < 2; ++mi)
      #pragma unroll
      for (int ni = 0; ni < 2; ++ni) {
        int n = nBase + ni * 16;
        float bv = bias ? bias[n] : 0.f;
        #pragma unroll
        for (int r = 0; r < 8; ++r) {
          int m = mBase + mi * 16 + r;
          O[(long)m * ldo + n] += acc[mi][ni][r] + bv;
        }
      }
  } else if (mode == MODE_SCORES) {
    float* O = (float*)OutV + (long)zb * oZ + (long)zh * oH;
    #pragma unroll
    for (int mi = 0; mi < 2; ++mi)
      #pragma unroll
      for (int ni = 0; ni < 2; ++ni) {
        int n = nBase + ni * 16;
        #pragma unroll
        for (int r = 0; r < 8; ++r) {
          int m = mBase + mi * 16 + r;
          float v = acc[mi][ni][r] * scale;
          O[(long)m * ldo + n] = (n <= m + maskOff) ? v : -1.0e30f;
        }
      }
  } else if (mode == MODE_VT) {
    __bf16* O = (__bf16*)OutV;
    #pragma unroll
    for (int mi = 0; mi < 2; ++mi)
      #pragma unroll
      for (int ni = 0; ni < 2; ++ni) {
        int n = nBase + ni * 16;
        float bv = bias ? bias[n] : 0.f;
        #pragma unroll
        for (int r = 0; r < 8; ++r) {
          int m = mBase + mi * 16 + r;        // global flattened (b*Srows + s)
          int bb = m / Srows, s = m - bb * Srows;
          O[(long)bb * vtBatch + (long)n * Srows + s] = (__bf16)(acc[mi][ni][r] + bv);
        }
      }
  } else {  // MODE_BF16 / MODE_SQRELU
    __bf16* O = (__bf16*)OutV + (long)zb * oZ + (long)zh * oH;
    #pragma unroll
    for (int mi = 0; mi < 2; ++mi)
      #pragma unroll
      for (int ni = 0; ni < 2; ++ni) {
        int n = nBase + ni * 16;
        float bv = bias ? bias[n] : 0.f;
        #pragma unroll
        for (int r = 0; r < 8; ++r) {
          int m = mBase + mi * 16 + r;
          float v = acc[mi][ni][r] + bv;
          if (mode == MODE_SQRELU) { v = v > 0.f ? v : 0.f; v = v * v; }
          O[(long)m * ldo + n] = (__bf16)v;
        }
      }
  }
}

// ---------------------------------------------------------------------------
__global__ void k_copy_out(const float* __restrict__ z, float* __restrict__ out, long n) {
  long i = (long)blockIdx.x * blockDim.x + threadIdx.x;
  if (i < n) out[i] = z[i];
}

// ===========================================================================
// Host orchestration
// ===========================================================================
static inline void launch_gemm(hipStream_t st, int M, int N, int K, int nz, int H,
                               const __bf16* A, long aZ, long aH, int lda,
                               const __bf16* Bm, long bZ, long bH, int ldb,
                               const float* bias, void* Out, long oZ, long oH, int ldo,
                               int mode, float scale = 0.f, int maskOff = 0,
                               int Srows = 1, long vtBatch = 0) {
  dim3 g(N / 64, M / 128, nz), b(256);
  k_gemm_wmma<<<g, b, 0, st>>>(A, aZ, aH, lda, Bm, bZ, bH, ldb, bias,
                               Out, oZ, oH, ldo, K, H, mode, scale, maskOff,
                               Srows, vtBatch);
}

extern "C" void kernel_launch(void* const* d_in, const int* in_sizes, int n_in,
                              void* d_out, int out_size, void* d_ws, size_t ws_size,
                              hipStream_t stream) {
  using namespace cfg;
  (void)in_sizes; (void)n_in; (void)out_size; (void)ws_size;

  const int*   inputs  = (const int*)d_in[0];
  const float* emb     = (const float*)d_in[1];
  const float* c_lnq_g = (const float*)d_in[2],  *c_lnq_b  = (const float*)d_in[3];
  const float* c_lnkv_g= (const float*)d_in[4],  *c_lnkv_b = (const float*)d_in[5];
  const float* cW[6]   = {(const float*)d_in[6], (const float*)d_in[8],
                          (const float*)d_in[10], (const float*)d_in[12],
                          (const float*)d_in[16], (const float*)d_in[18]};
  const float* cBias[6]= {(const float*)d_in[7], (const float*)d_in[9],
                          (const float*)d_in[11], (const float*)d_in[13],
                          (const float*)d_in[17], (const float*)d_in[19]};
  const float* c_ln2_g = (const float*)d_in[14], *c_ln2_b = (const float*)d_in[15];
  const float* s_lnq_g = (const float*)d_in[20], *s_lnq_b = (const float*)d_in[21];
  const float* sW[6]   = {(const float*)d_in[22], (const float*)d_in[24],
                          (const float*)d_in[26], (const float*)d_in[28],
                          (const float*)d_in[32], (const float*)d_in[34]};
  const float* sBias[6]= {(const float*)d_in[23], (const float*)d_in[25],
                          (const float*)d_in[27], (const float*)d_in[29],
                          (const float*)d_in[33], (const float*)d_in[35]};
  const float* s_ln2_g = (const float*)d_in[30], *s_ln2_b = (const float*)d_in[31];

  char* ws = (char*)d_ws;

  // Workspace layout (bytes, all naturally 256B aligned)
  const size_t WT_OFF = 0;                                      // 18 x C*C bf16
  const size_t EN_OFF = WT_OFF + 18UL * C * C * 2;              // B*S*C bf16
  const size_t Z_OFF  = EN_OFF + (size_t)B * S * C * 2;         // B*Z*C f32
  const size_t ZN_OFF = Z_OFF  + (size_t)B * Z * C * 4;         // B*Z*C bf16
  const size_t QB_OFF = ZN_OFF + (size_t)B * Z * C * 2;
  const size_t KB_OFF = QB_OFF + (size_t)B * Z * C * 2;         // B*S*C bf16
  const size_t VT_OFF = KB_OFF + (size_t)B * S * C * 2;         // B*C*S bf16 (transposed V)
  const size_t AT_OFF = VT_OFF + (size_t)B * C * S * 2;         // B*Z*C bf16
  const size_t HB_OFF = AT_OFF + (size_t)B * Z * C * 2;         // B*Z*C bf16
  const size_t SC_OFF = HB_OFF + (size_t)B * Z * C * 2;         // max(B*Z*S, B*HS*Z*Z) f32
  const size_t P_OFF  = SC_OFF + (size_t)B * Z * S * 4;         // same count, bf16

  __bf16* wt0   = (__bf16*)(ws + WT_OFF);
  __bf16* en    = (__bf16*)(ws + EN_OFF);
  float*  zf    = (float*) (ws + Z_OFF);
  __bf16* znb   = (__bf16*)(ws + ZN_OFF);
  __bf16* qb    = (__bf16*)(ws + QB_OFF);
  __bf16* kb    = (__bf16*)(ws + KB_OFF);
  __bf16* vt    = (__bf16*)(ws + VT_OFF);
  __bf16* attno = (__bf16*)(ws + AT_OFF);
  __bf16* hb    = (__bf16*)(ws + HB_OFF);
  float*  scf   = (float*) (ws + SC_OFF);
  __bf16* pb    = (__bf16*)(ws + P_OFF);

  auto WT = [&](int i) { return wt0 + (size_t)i * C * C; };  // 0..5 cross, 6+l*6+i self

  // --- 1) weight transpose+convert ---
  {
    dim3 g((C * C) / 256), b(256);
    for (int i = 0; i < 6; ++i)
      k_transpose_bf16<<<g, b, 0, stream>>>(cW[i], WT(i), C, C);
    for (int l = 0; l < L; ++l)
      for (int i = 0; i < 6; ++i)
        k_transpose_bf16<<<g, b, 0, stream>>>(sW[i] + (size_t)l * C * C, WT(6 + l * 6 + i), C, C);
  }

  // --- 2) gather + LayerNorms ---
  k_gather_ln<<<B * S, 256, 0, stream>>>(inputs, emb, c_lnkv_g, c_lnkv_b,
                                         c_lnq_g, c_lnq_b, en, zf, znb);

  const float crossScale = 1.0f / 32.0f;          // 1/sqrt(1024)
  const float selfScale  = 0.0883883476483f;      // 1/sqrt(128)
  const int   d = C / HS;                         // 128

  // --- 3) cross attention (HC = 1, d = C) ---
  launch_gemm(stream, B * Z, C, C, 1, 1, znb, 0, 0, C, WT(0), 0, 0, C,
              cBias[0], qb, 0, 0, C, MODE_BF16);
  launch_gemm(stream, B * S, C, C, 1, 1, en, 0, 0, C, WT(1), 0, 0, C,
              cBias[1], kb, 0, 0, C, MODE_BF16);
  launch_gemm(stream, B * S, C, C, 1, 1, en, 0, 0, C, WT(2), 0, 0, C,
              cBias[2], vt, 0, 0, 0, MODE_VT, 0.f, 0, S, (long)C * S);
  // scores[b][q][k], mask: k <= q + (S - Z)
  launch_gemm(stream, Z, S, C, B, 1,
              qb, (long)Z * C, 0, C, kb, (long)S * C, 0, C, nullptr,
              scf, (long)Z * S, 0, S, MODE_SCORES, crossScale, S - Z);
  k_softmax<<<B * Z, 256, 0, stream>>>(scf, pb, S);
  launch_gemm(stream, Z, C, S, B, 1,
              pb, (long)Z * S, 0, S, vt, (long)C * S, 0, S, nullptr,
              attno, (long)Z * C, 0, C, MODE_BF16);
  launch_gemm(stream, B * Z, C, C, 1, 1, attno, 0, 0, C, WT(3), 0, 0, C,
              cBias[3], zf, 0, 0, C, MODE_F32ADD);
  // cross MLP
  k_ln<<<B * Z, 256, 0, stream>>>(zf, c_ln2_g, c_ln2_b, znb);
  launch_gemm(stream, B * Z, C, C, 1, 1, znb, 0, 0, C, WT(4), 0, 0, C,
              cBias[4], hb, 0, 0, C, MODE_SQRELU);
  launch_gemm(stream, B * Z, C, C, 1, 1, hb, 0, 0, C, WT(5), 0, 0, C,
              cBias[5], zf, 0, 0, C, MODE_F32ADD);

  // --- 4) self attention blocks ---
  for (int nb = 0; nb < NB; ++nb) {
    for (int l = 0; l < L; ++l) {
      const int wq = 6 + l * 6;
      k_ln<<<B * Z, 256, 0, stream>>>(zf, s_lnq_g + l * C, s_lnq_b + l * C, znb);
      launch_gemm(stream, B * Z, C, C, 1, 1, znb, 0, 0, C, WT(wq + 0), 0, 0, C,
                  sBias[0] + l * C, qb, 0, 0, C, MODE_BF16);
      launch_gemm(stream, B * Z, C, C, 1, 1, znb, 0, 0, C, WT(wq + 1), 0, 0, C,
                  sBias[1] + l * C, kb, 0, 0, C, MODE_BF16);
      launch_gemm(stream, B * Z, C, C, 1, 1, znb, 0, 0, C, WT(wq + 2), 0, 0, C,
                  sBias[2] + l * C, vt, 0, 0, 0, MODE_VT, 0.f, 0, Z, (long)C * Z);
      // scores[b][h][q][k], causal mask, d = 128
      launch_gemm(stream, Z, Z, d, B * HS, HS,
                  qb, (long)Z * C, d, C, kb, (long)Z * C, d, C, nullptr,
                  scf, (long)HS * Z * Z, (long)Z * Z, Z, MODE_SCORES, selfScale, 0);
      k_softmax<<<B * HS * Z, 256, 0, stream>>>(scf, pb, Z);
      // o[b][q][h*d + c] = P @ Vt
      launch_gemm(stream, Z, d, Z, B * HS, HS,
                  pb, (long)HS * Z * Z, (long)Z * Z, Z,
                  vt, (long)C * Z, (long)d * Z, Z, nullptr,
                  attno, (long)Z * C, d, C, MODE_BF16);
      launch_gemm(stream, B * Z, C, C, 1, 1, attno, 0, 0, C, WT(wq + 3), 0, 0, C,
                  sBias[3] + l * C, zf, 0, 0, C, MODE_F32ADD);
      // MLP
      k_ln<<<B * Z, 256, 0, stream>>>(zf, s_ln2_g + l * C, s_ln2_b + l * C, znb);
      launch_gemm(stream, B * Z, C, C, 1, 1, znb, 0, 0, C, WT(wq + 4), 0, 0, C,
                  sBias[4] + l * C, hb, 0, 0, C, MODE_SQRELU);
      launch_gemm(stream, B * Z, C, C, 1, 1, hb, 0, 0, C, WT(wq + 5), 0, 0, C,
                  sBias[5] + l * C, zf, 0, 0, C, MODE_F32ADD);
    }
  }

  // --- 5) output ---
  long nOut = (long)B * Z * C;
  k_copy_out<<<(unsigned)((nOut + 255) / 256), 256, 0, stream>>>(zf, (float*)d_out, nOut);
}
